// GNN_15951508537738
// MI455X (gfx1250) — matrix operand order repaired and checked by reading, not compiled, not run on hardware
//
#include <hip/hip_runtime.h>

// ---------------------------------------------------------------------------
// Problem constants (from reference setup_inputs)
// ---------------------------------------------------------------------------
#define BSZ   8
#define NEN   48
#define DE    768      // ent / rel feature dim
#define DM    512      // mem dim
#define ME    (BSZ*NEN)          // 384  entity rows
#define MR    (BSZ*NEN*NEN)      // 18432 relation rows
#define NEGV  (-10000.0f)

typedef __attribute__((ext_vector_type(16))) __bf16 v16bf;
typedef __attribute__((ext_vector_type(8)))  __bf16 v8bf;
typedef __attribute__((ext_vector_type(8)))  float  v8f;

// ---------------------------------------------------------------------------
// Helpers
// ---------------------------------------------------------------------------
__device__ __forceinline__ float gelu_f(float x) {
    return 0.5f * x * (1.0f + erff(x * 0.70710678118654752440f));
}

__device__ __forceinline__ float wave_reduce_add(float p) {
    #pragma unroll
    for (int o = 16; o > 0; o >>= 1) p += __shfl_down(p, o, 32);
    return p;
}

// ---------------------------------------------------------------------------
// f32 -> bf16 convert
// ---------------------------------------------------------------------------
__global__ void cvt_bf16_kernel(const float* __restrict__ s,
                                __bf16* __restrict__ d, int n) {
    int i = blockIdx.x * blockDim.x + threadIdx.x;
    if (i < n) d[i] = (__bf16)s[i];
}

// Transposed convert: Wt[k*N + n] = W[n*ldw + colOff + k]   (Wt is K x N, bf16)
__global__ void tcvt_bf16_kernel(const float* __restrict__ W,
                                 __bf16* __restrict__ Wt,
                                 int N, int K, int ldw, int colOff) {
    int idx = blockIdx.x * blockDim.x + threadIdx.x;
    if (idx >= N * K) return;
    int k = idx / N, n = idx % N;
    Wt[idx] = (__bf16)W[(size_t)n * ldw + colOff + k];
}

// ---------------------------------------------------------------------------
// WMMA GEMM:  C[M,N] = A[M,K](bf16,row-major) * Bt[K,N](bf16,row-major) + bias
//
// Register blocking: one wave -> 32x64 of C (2 M-subtiles x 4 N-subtiles,
// 8 f32 accumulators). Each B fragment is reused by 2 WMMAs; per 32-K step:
// 8 WMMAs (128K FLOP) vs 4 x b128 A + 64 x b16 B loads/lane (6 KB/wave) -> ~21 FLOP/B from cache.
// A block's 8 waves share one 64-wide N strip (same B cachelines), covering
// a 256x64 C tile; grid = (N/64, ceil(M/256)).
//
// Fragment layouts per CDNA5 ISA 7.12.2 (wave32):
//   A 16x32 bf16 : lane%16 = M row, lane/16 selects K-chunks {0..7,16..23}
//                  vs {8..15,24..31}; two contiguous 8-elem (16B) loads.
//   B 32x16 bf16 : lane%16 = N col, K-chunks by lane/16 as for A; 16 strided 2B loads.
//   C/D 16x16 f32: VGPR r -> M = r + 8*(lane/16); N = lane%16.
// ---------------------------------------------------------------------------
__global__ __launch_bounds__(256)
void wmma_gemm_bf16(const __bf16* __restrict__ A,
                    const __bf16* __restrict__ Bt,
                    const float* __restrict__ bias,
                    float* __restrict__ C,
                    int M, int N, int K) {
    const int wave = threadIdx.x >> 5;
    const int lane = threadIdx.x & 31;
    const int n0 = blockIdx.x << 6;
    const int m0 = (blockIdx.y * 8 + wave) << 5;     // 32 rows per wave
    if (m0 >= M) return;

    const int half = lane >> 4;      // which K half-chunk set
    const int mr   = lane & 15;      // M row within A fragment / N col in C

    v8f accA0 = {}, accA1 = {}, accA2 = {}, accA3 = {};   // rows m0..m0+15
    v8f accB0 = {}, accB1 = {}, accB2 = {}, accB3 = {};   // rows m0+16..m0+31

    const __bf16* arow0 = A + (size_t)(m0 + mr) * K + half * 8;
    const __bf16* arow1 = arow0 + (size_t)16 * K;

    for (int k0 = 0; k0 < K; k0 += 32) {
        v8bf a0lo = *(const v8bf*)(arow0 + k0);        // K = 8*half + 0..7
        v8bf a0hi = *(const v8bf*)(arow0 + k0 + 16);   // K = 16 + 8*half + 0..7
        v8bf a1lo = *(const v8bf*)(arow1 + k0);
        v8bf a1hi = *(const v8bf*)(arow1 + k0 + 16);
        v16bf a0 = __builtin_shufflevector(a0lo, a0hi,
                     0,1,2,3,4,5,6,7,8,9,10,11,12,13,14,15);
        v16bf a1 = __builtin_shufflevector(a1lo, a1hi,
                     0,1,2,3,4,5,6,7,8,9,10,11,12,13,14,15);

        const __bf16* brow = Bt + (size_t)(k0 + lane) * N + n0;
        __builtin_prefetch((const void*)(brow + 32 * N), 0, 0);
        const __bf16* bcol = Bt + (size_t)k0 * N + n0 + mr;
        v16bf b0, b1, b2, b3;
        #pragma unroll
        for (int s = 0; s < 16; ++s) {
            int kk = (s < 8) ? (8 * half + s) : (16 + 8 * half + (s - 8));
            const __bf16* bp = bcol + (size_t)kk * N;
            b0[s] = bp[0];
            b1[s] = bp[16];
            b2[s] = bp[32];
            b3[s] = bp[48];
        }

        accA0 = __builtin_amdgcn_wmma_f32_16x16x32_bf16(false, a0, false, b0, (short)0, accA0, false, false);
        accB0 = __builtin_amdgcn_wmma_f32_16x16x32_bf16(false, a1, false, b0, (short)0, accB0, false, false);
        accA1 = __builtin_amdgcn_wmma_f32_16x16x32_bf16(false, a0, false, b1, (short)0, accA1, false, false);
        accB1 = __builtin_amdgcn_wmma_f32_16x16x32_bf16(false, a1, false, b1, (short)0, accB1, false, false);
        accA2 = __builtin_amdgcn_wmma_f32_16x16x32_bf16(false, a0, false, b2, (short)0, accA2, false, false);
        accB2 = __builtin_amdgcn_wmma_f32_16x16x32_bf16(false, a1, false, b2, (short)0, accB2, false, false);
        accA3 = __builtin_amdgcn_wmma_f32_16x16x32_bf16(false, a0, false, b3, (short)0, accA3, false, false);
        accB3 = __builtin_amdgcn_wmma_f32_16x16x32_bf16(false, a1, false, b3, (short)0, accB3, false, false);
    }

    const int col = mr;
    float bv0 = bias ? bias[n0 + col]      : 0.0f;
    float bv1 = bias ? bias[n0 + col + 16] : 0.0f;
    float bv2 = bias ? bias[n0 + col + 32] : 0.0f;
    float bv3 = bias ? bias[n0 + col + 48] : 0.0f;

    #pragma unroll
    for (int r = 0; r < 8; ++r) {
        int row0 = m0 + r + 8 * half;
        float* c0 = C + (size_t)row0 * N + n0 + col;
        c0[0]  = accA0[r] + bv0;
        c0[16] = accA1[r] + bv1;
        c0[32] = accA2[r] + bv2;
        c0[48] = accA3[r] + bv3;
        float* c1 = c0 + (size_t)16 * N;
        c1[0]  = accB0[r] + bv0;
        c1[16] = accB1[r] + bv1;
        c1[32] = accB2[r] + bv2;
        c1[48] = accB3[r] + bv3;
    }
}

// ---------------------------------------------------------------------------
// Fused energy / masked-softmax / weighted-sum for sub & obj updates.
// One block per (b, p): p = subject i (objMode=0) or object j (objMode=1).
// 49 slots: slot 0 = self h, slots 1..48 = neighbors via rel-derived h.
// energy_k = wv . gelu(zA + zB_k)   (biases folded into zA GEMM)
// ---------------------------------------------------------------------------
__global__ __launch_bounds__(256)
void attn_so_kernel(const float* __restrict__ zA,
                    const float* __restrict__ zBs,
                    const float* __restrict__ zBn,
                    const float* __restrict__ hs,
                    const float* __restrict__ hn,
                    const float* __restrict__ wv,
                    const int* __restrict__ ent,
                    float* __restrict__ out,
                    int ne, int objMode) {
    const int blk = blockIdx.x;
    const int b = blk / ne, p = blk % ne;
    const int nb = ent[b];
    const size_t selfOff  = (size_t)blk * DM;
    const size_t nbBase   = objMode ? ((size_t)b * ne * ne + p) * DM
                                    : (size_t)blk * ne * DM;
    const size_t nbStride = objMode ? (size_t)ne * DM : (size_t)DM;
    const int wave = threadIdx.x >> 5, lane = threadIdx.x & 31;
    const int nw = blockDim.x >> 5;

    __shared__ float eng[64];

    const float* za = zA + selfOff;
    for (int k = wave; k <= ne; k += nw) {
        const float* zb = (k == 0) ? (zBs + selfOff)
                                   : (zBn + nbBase + (size_t)(k - 1) * nbStride);
        float pch = 0.0f;
        for (int d = lane; d < DM; d += 32)
            pch += wv[d] * gelu_f(za[d] + zb[d]);
        pch = wave_reduce_add(pch);
        if (lane == 0) {
            if (k > 0) {
                int t = k - 1;
                if (t == p || t >= nb || p >= nb) pch = NEGV;
            }
            eng[k] = pch;
        }
    }
    __syncthreads();

    if (threadIdx.x == 0) {
        float mx = -3.0e38f;
        for (int k = 0; k <= ne; ++k) mx = fmaxf(mx, eng[k]);
        float s = 0.0f;
        for (int k = 0; k <= ne; ++k) { float v = __expf(eng[k] - mx); eng[k] = v; s += v; }
        float inv = 1.0f / s;
        for (int k = 0; k <= ne; ++k) eng[k] *= inv;
    }
    __syncthreads();

    for (int d = threadIdx.x; d < DM; d += blockDim.x) {
        float acc = eng[0] * hs[selfOff + d];
        for (int t = 0; t < ne; ++t)
            acc += eng[t + 1] * hn[nbBase + (size_t)t * nbStride + d];
        out[selfOff + d] = acc;
    }
}

// ---------------------------------------------------------------------------
// Fused 3-slot attention for the relation update. Out written in-place to h0.
// ---------------------------------------------------------------------------
__global__ __launch_bounds__(256)
void attn_rel_kernel(const float* __restrict__ zA,
                     const float* __restrict__ zB0,
                     const float* __restrict__ zB1,
                     const float* __restrict__ zB2,
                     float* __restrict__ h0,      // in: hr, out: attn sum
                     const float* __restrict__ h1,
                     const float* __restrict__ h2,
                     const float* __restrict__ wv,
                     int ne) {
    const int row = blockIdx.x;               // (b*ne + i)*ne + j
    const int j = row % ne;
    const int bi = row / ne;
    const int i = bi % ne;
    const int b = bi / ne;
    const size_t o0 = (size_t)row * DM;
    const size_t o1 = ((size_t)b * ne + i) * DM;
    const size_t o2 = ((size_t)b * ne + j) * DM;
    const int wave = threadIdx.x >> 5, lane = threadIdx.x & 31;

    __shared__ float eng[3];

    if (wave < 3) {
        const float* zb = (wave == 0) ? (zB0 + o0) : (wave == 1 ? (zB1 + o1) : (zB2 + o2));
        const float* za = zA + o0;
        float p = 0.0f;
        for (int d = lane; d < DM; d += 32)
            p += wv[d] * gelu_f(za[d] + zb[d]);
        p = wave_reduce_add(p);
        if (lane == 0) eng[wave] = p;
    }
    __syncthreads();
    if (threadIdx.x == 0) {
        float mx = fmaxf(eng[0], fmaxf(eng[1], eng[2]));
        float e0 = __expf(eng[0] - mx), e1 = __expf(eng[1] - mx), e2 = __expf(eng[2] - mx);
        float inv = 1.0f / (e0 + e1 + e2);
        eng[0] = e0 * inv; eng[1] = e1 * inv; eng[2] = e2 * inv;
    }
    __syncthreads();
    for (int d = threadIdx.x; d < DM; d += blockDim.x) {
        float a = eng[0] * h0[o0 + d] + eng[1] * h1[o1 + d] + eng[2] * h2[o2 + d];
        h0[o0 + d] = a;     // same thread read-then-write: safe
    }
}

// ---------------------------------------------------------------------------
// LayerNorm(fc + res) * mask -> dst.   mode 1: mask1d rows; mode 2: mask2d.
// ---------------------------------------------------------------------------
__global__ __launch_bounds__(256)
void ln_mask_kernel(const float* __restrict__ fc,
                    const float* __restrict__ res,
                    const float* __restrict__ g,
                    const float* __restrict__ bta,
                    const int* __restrict__ ent,
                    float* __restrict__ dst,
                    int D, int ne, int mode) {
    const int row = blockIdx.x;
    int valid;
    if (mode == 1) {
        int b = row / ne, i = row % ne;
        valid = (i < ent[b]);
    } else {
        int j = row % ne; int t = row / ne; int i = t % ne; int b = t / ne;
        int nb = ent[b];
        valid = (i < nb) && (j < nb);
    }
    const float* x0 = fc + (size_t)row * D;
    const float* r0 = res + (size_t)row * D;

    float s = 0.0f, s2 = 0.0f;
    for (int d = threadIdx.x; d < D; d += blockDim.x) {
        float v = x0[d] + r0[d];
        s += v; s2 += v * v;
    }
    __shared__ float rs[256], rq[256];
    rs[threadIdx.x] = s; rq[threadIdx.x] = s2;
    __syncthreads();
    for (int o = 128; o > 0; o >>= 1) {
        if ((int)threadIdx.x < o) {
            rs[threadIdx.x] += rs[threadIdx.x + o];
            rq[threadIdx.x] += rq[threadIdx.x + o];
        }
        __syncthreads();
    }
    float mean = rs[0] / D;
    float var  = rq[0] / D - mean * mean;
    float inv  = rsqrtf(var + 1e-6f);
    float m    = valid ? 1.0f : 0.0f;
    for (int d = threadIdx.x; d < D; d += blockDim.x) {
        float v = x0[d] + r0[d];
        dst[(size_t)row * D + d] = ((v - mean) * inv * g[d] + bta[d]) * m;
    }
}

// ---------------------------------------------------------------------------
// Host orchestration
// ---------------------------------------------------------------------------
extern "C" void kernel_launch(void* const* d_in, const int* in_sizes, int n_in,
                              void* d_out, int out_size, void* d_ws, size_t ws_size,
                              hipStream_t stream) {
    (void)in_sizes; (void)n_in; (void)out_size; (void)ws_size;

    const float* sub_in = (const float*)d_in[0];
    const float* obj_in = (const float*)d_in[1];
    const float* rel_in = (const float*)d_in[2];
    const int*   ent    = (const int*)d_in[3];
    auto P = [&](int i) { return (const float*)d_in[i]; };
    // param indices (dict insertion order, offset +4):
    // 4 W_kv_s 5 b_kv_s 6 W_kv_o 7 b_kv_o 8 W_kv_rs 9 b_kv_rs 10 W_kv_ro 11 b_kv_ro
    // 12 W_comb_s 13 b_comb_s 14 W_comb_o 15 b_comb_o 16 w_sv 17 w_ov
    // 18 W_fc_s 19 b_fc_s 20 W_fc_o 21 b_fc_o 22 W_kv_r 23 b_kv_r 24 W_kv_sr 25 b_kv_sr
    // 26 W_kv_or 27 b_kv_or 28 W_comb_r 29 b_comb_r 30 w_rv 31 W_fc_r 32 b_fc_r
    // 33 ln_s_g 34 ln_s_b 35 ln_o_g 36 ln_o_b 37 ln_r_g 38 ln_r_b

    // ----- workspace arena -----
    char* wp = (char*)d_ws;
    auto alloc = [&](size_t bytes) -> void* {
        void* r = (void*)wp;
        wp += (bytes + 255) & ~(size_t)255;
        return r;
    };
    // transposed bf16 weights (K x N)
    __bf16* Wt_kv_s  = (__bf16*)alloc((size_t)DE * DM * 2);
    __bf16* Wt_kv_o  = (__bf16*)alloc((size_t)DE * DM * 2);
    __bf16* Wt_kv_rs = (__bf16*)alloc((size_t)DE * DM * 2);
    __bf16* Wt_kv_ro = (__bf16*)alloc((size_t)DE * DM * 2);
    __bf16* Wt_kv_r  = (__bf16*)alloc((size_t)DE * DM * 2);
    __bf16* Wt_kv_sr = (__bf16*)alloc((size_t)DE * DM * 2);
    __bf16* Wt_kv_or = (__bf16*)alloc((size_t)DE * DM * 2);
    __bf16* Wt1_s    = (__bf16*)alloc((size_t)DE * DM * 2);
    __bf16* Wt1_o    = (__bf16*)alloc((size_t)DE * DM * 2);
    __bf16* Wt1_r    = (__bf16*)alloc((size_t)DE * DM * 2);
    __bf16* Wt2_s    = (__bf16*)alloc((size_t)DM * DM * 2);
    __bf16* Wt2_o    = (__bf16*)alloc((size_t)DM * DM * 2);
    __bf16* Wt2_r    = (__bf16*)alloc((size_t)DM * DM * 2);
    __bf16* Wt_fc_s  = (__bf16*)alloc((size_t)DM * DE * 2);
    __bf16* Wt_fc_o  = (__bf16*)alloc((size_t)DM * DE * 2);
    __bf16* Wt_fc_r  = (__bf16*)alloc((size_t)DM * DE * 2);
    // bf16 activations
    __bf16* subbf = (__bf16*)alloc((size_t)ME * DE * 2);
    __bf16* objbf = (__bf16*)alloc((size_t)ME * DE * 2);
    __bf16* relbf = (__bf16*)alloc((size_t)MR * DE * 2);
    // small f32 scratch (entity-row sized)
    float* hsF    = (float*)alloc((size_t)ME * DM * 4);
    float* hoF    = (float*)alloc((size_t)ME * DM * 4);
    float* zA_sm  = (float*)alloc((size_t)ME * DM * 4);
    float* zBs_sm = (float*)alloc((size_t)ME * DM * 4);
    float* zBo_sm = (float*)alloc((size_t)ME * DM * 4);
    float* out_sm = (float*)alloc((size_t)ME * DM * 4);
    __bf16* smbf  = (__bf16*)alloc((size_t)ME * DM * 2);
    __bf16* smbf2 = (__bf16*)alloc((size_t)ME * DM * 2);
    // big scratch (relation-row sized)
    float*  S1 = (float*)alloc((size_t)MR * DM * 4);   // h of rel (also attn out in rel phase)
    float*  S2 = (float*)alloc((size_t)MR * DM * 4);   // zB of rel-h
    float*  S3 = (float*)alloc((size_t)MR * DM * 4);   // zA of rel phase
    __bf16* S4 = (__bf16*)alloc((size_t)MR * DM * 2);  // bf16 staging
    float*  S5 = (float*)alloc((size_t)MR * DE * 4);   // fc output (max size)
    // double-buffered activations for iteration 0 -> 1
    float* sub1 = (float*)alloc((size_t)ME * DE * 4);
    float* obj1 = (float*)alloc((size_t)ME * DE * 4);
    float* rel1 = (float*)alloc((size_t)MR * DE * 4);

    // ----- launch helpers -----
    auto gemm = [&](const __bf16* A, const __bf16* Bt, const float* bias,
                    float* C, int M, int N, int K) {
        dim3 grid(N / 64, (M + 255) / 256);
        wmma_gemm_bf16<<<grid, 256, 0, stream>>>(A, Bt, bias, C, M, N, K);
    };
    auto cvt = [&](const float* s, __bf16* d, int n) {
        cvt_bf16_kernel<<<(n + 255) / 256, 256, 0, stream>>>(s, d, n);
    };
    auto tcvt = [&](const float* W, __bf16* Wt, int N, int K, int ldw, int off) {
        int n = N * K;
        tcvt_bf16_kernel<<<(n + 255) / 256, 256, 0, stream>>>(W, Wt, N, K, ldw, off);
    };

    // ----- weight prep (bf16, K-major) -----
    tcvt(P(4),  Wt_kv_s,  DM, DE, DE, 0);
    tcvt(P(6),  Wt_kv_o,  DM, DE, DE, 0);
    tcvt(P(8),  Wt_kv_rs, DM, DE, DE, 0);
    tcvt(P(10), Wt_kv_ro, DM, DE, DE, 0);
    tcvt(P(22), Wt_kv_r,  DM, DE, DE, 0);
    tcvt(P(24), Wt_kv_sr, DM, DE, DE, 0);
    tcvt(P(26), Wt_kv_or, DM, DE, DE, 0);
    tcvt(P(12), Wt1_s, DM, DE, DM + DE, 0);    // W_comb_s cols [0,768)
    tcvt(P(12), Wt2_s, DM, DM, DM + DE, DE);   // W_comb_s cols [768,1280)
    tcvt(P(14), Wt1_o, DM, DE, DM + DE, 0);
    tcvt(P(14), Wt2_o, DM, DM, DM + DE, DE);
    tcvt(P(28), Wt1_r, DM, DE, DM + DE, 0);
    tcvt(P(28), Wt2_r, DM, DM, DM + DE, DE);
    tcvt(P(18), Wt_fc_s, DE, DM, DM, 0);
    tcvt(P(20), Wt_fc_o, DE, DM, DM, 0);
    tcvt(P(31), Wt_fc_r, DE, DM, DM, 0);

    const float* sub_cur = sub_in;
    const float* obj_cur = obj_in;
    const float* rel_cur = rel_in;
    float* outp = (float*)d_out;

    for (int it = 0; it < 2; ++it) {
        float* sub_dst = (it == 0) ? sub1 : outp;
        float* obj_dst = (it == 0) ? obj1 : (outp + (size_t)ME * DE);
        float* rel_dst = (it == 0) ? rel1 : (outp + (size_t)2 * ME * DE);

        cvt(sub_cur, subbf, ME * DE);
        cvt(obj_cur, objbf, ME * DE);
        cvt(rel_cur, relbf, MR * DE);

        // ================= subject update =================
        gemm(subbf, Wt_kv_s,  P(5),  hsF,   ME, DM, DE);   // hs
        gemm(relbf, Wt_kv_rs, P(9),  S1,    MR, DM, DE);   // hr (per i,j)
        gemm(subbf, Wt1_s,    P(13), zA_sm, ME, DM, DE);   // zA (+b_comb)
        cvt(hsF, smbf, ME * DM);
        gemm(smbf, Wt2_s, nullptr, zBs_sm, ME, DM, DM);    // zB self
        cvt(S1, S4, MR * DM);
        gemm(S4, Wt2_s, nullptr, S2, MR, DM, DM);          // zB neighbors
        attn_so_kernel<<<ME, 256, 0, stream>>>(zA_sm, zBs_sm, S2, hsF, S1,
                                               P(16), ent, out_sm, NEN, 0);
        cvt(out_sm, smbf, ME * DM);
        gemm(smbf, Wt_fc_s, P(19), S5, ME, DE, DM);
        ln_mask_kernel<<<ME, 256, 0, stream>>>(S5, sub_cur, P(33), P(34),
                                               ent, sub_dst, DE, NEN, 1);

        // ================= object update =================
        gemm(objbf, Wt_kv_o,  P(7),  hsF,   ME, DM, DE);   // ho
        gemm(relbf, Wt_kv_ro, P(11), S1,    MR, DM, DE);   // hr
        gemm(objbf, Wt1_o,    P(15), zA_sm, ME, DM, DE);
        cvt(hsF, smbf, ME * DM);
        gemm(smbf, Wt2_o, nullptr, zBs_sm, ME, DM, DM);
        cvt(S1, S4, MR * DM);
        gemm(S4, Wt2_o, nullptr, S2, MR, DM, DM);
        attn_so_kernel<<<ME, 256, 0, stream>>>(zA_sm, zBs_sm, S2, hsF, S1,
                                               P(17), ent, out_sm, NEN, 1);
        cvt(out_sm, smbf, ME * DM);
        gemm(smbf, Wt_fc_o, P(21), S5, ME, DE, DM);
        ln_mask_kernel<<<ME, 256, 0, stream>>>(S5, obj_cur, P(35), P(36),
                                               ent, obj_dst, DE, NEN, 1);

        // ================= relation update =================
        gemm(relbf, Wt_kv_r,  P(23), S1,  MR, DM, DE);     // hr
        gemm(subbf, Wt_kv_sr, P(25), hsF, ME, DM, DE);     // hs_r
        gemm(objbf, Wt_kv_or, P(27), hoF, ME, DM, DE);     // ho_r
        gemm(relbf, Wt1_r,    P(29), S3,  MR, DM, DE);     // zA (+b_comb_r)
        cvt(S1, S4, MR * DM);
        gemm(S4, Wt2_r, nullptr, S2, MR, DM, DM);          // zB of hr
        cvt(hsF, smbf, ME * DM);
        gemm(smbf, Wt2_r, nullptr, zBs_sm, ME, DM, DM);    // zB of hs_r
        cvt(hoF, smbf2, ME * DM);
        gemm(smbf2, Wt2_r, nullptr, zBo_sm, ME, DM, DM);   // zB of ho_r
        attn_rel_kernel<<<MR, 256, 0, stream>>>(S3, S2, zBs_sm, zBo_sm,
                                                S1, hsF, hoF, P(30), NEN);
        cvt(S1, S4, MR * DM);                              // attn out -> bf16
        gemm(S4, Wt_fc_r, P(32), S5, MR, DE, DM);
        ln_mask_kernel<<<MR, 256, 0, stream>>>(S5, rel_cur, P(37), P(38),
                                               ent, rel_dst, DE, NEN, 2);

        sub_cur = sub_dst;
        obj_cur = obj_dst;
        rel_cur = rel_dst;
    }
}
